// MultiPositiveContrastiveLoss_23356032155982
// MI455X (gfx1250) — compile-verified, matrix operand order checked by reading, not run on hardware
//
#include <hip/hip_runtime.h>

// ---- problem constants (from reference) ----
#define BB   32
#define CC   256
#define NNX  64
#define NN2  (NNX*NNX)     // 4096
#define SS   64            // B*NUM_SENT
#define MM   128           // S*NUM_TGT (K=1)
#define PP   2080          // N(N+1)/2 upper-tri proposals
#define BP   (BB*PP)       // 66560
#define PW   (BP/32)       // 2080 mask words per row
#define EE   256           // S * (NUM_TGT*K)^2
#define NEGS 4096
#define INV_T        10.0f // 1/T
#define INTER_MARGIN 0.3f
#define COLTILES (BP/64)   // 1040
#define QROWS 192          // 64 sf rows + 128 pvf rows

typedef __attribute__((ext_vector_type(8)))  unsigned short v8u;
typedef __attribute__((ext_vector_type(16))) unsigned short v16u;
typedef __attribute__((ext_vector_type(16))) __bf16        v16bf;
typedef __attribute__((ext_vector_type(8)))  float         v8f;
typedef __attribute__((ext_vector_type(4)))  int           v4i;

// ---- CDNA5 async global->LDS staging (ASYNCcnt path), with safe fallback ----
#if defined(__has_builtin)
#if __has_builtin(__builtin_amdgcn_global_load_async_to_lds_b128) && \
    __has_builtin(__builtin_amdgcn_s_wait_asynccnt)
#define USE_ASYNC_LDS 1
#endif
#endif
#ifndef USE_ASYNC_LDS
#define USE_ASYNC_LDS 0
#endif

#if USE_ASYNC_LDS
typedef __attribute__((address_space(1))) v4i as1_v4i;   // global int4*
typedef __attribute__((address_space(3))) v4i as3_v4i;   // LDS int4*
__device__ __forceinline__ void async_copy16(const void* gsrc, void* ldst) {
  // generic->LDS: LDS offset lives in the low 32 bits of the generic address
  __builtin_amdgcn_global_load_async_to_lds_b128(
      (as1_v4i*)(uintptr_t)gsrc,
      (as3_v4i*)(unsigned int)(uintptr_t)ldst,
      0, 0);
}
__device__ __forceinline__ void async_wait0() {
  __builtin_amdgcn_s_wait_asynccnt(0);
}
#endif

__device__ __forceinline__ unsigned short f2bf(float x) {
  unsigned int u = __builtin_bit_cast(unsigned int, x);
  unsigned int r = (u + 0x7FFFu + ((u >> 16) & 1u)) >> 16;
  return (unsigned short)r;
}

__device__ __forceinline__ float u01(unsigned int a, unsigned int b, unsigned int c) {
  unsigned int h = a * 0x9E3779B1u ^ b * 0x85EBCA77u ^ c * 0xC2B2AE3Du;
  h ^= h >> 16; h *= 0x7FEB352Du; h ^= h >> 15; h *= 0x846CA68Bu; h ^= h >> 16;
  return (float)(h >> 8) * (1.0f / 16777216.0f);
}

// Load a 16xK bf16 WMMA fragment from LDS.
// koff = kk*32 + (lane>>4)*8 per ISA 16-bit A/B layout (two 16B chunks, +16 apart).
__device__ __forceinline__ v16bf load_frag(const unsigned short* rowbase, int koff) {
  v8u lo = *(const v8u*)(rowbase + koff);
  v8u hi = *(const v8u*)(rowbase + koff + 16);
  v16u full = __builtin_shufflevector(lo, hi, 0,1,2,3,4,5,6,7,8,9,10,11,12,13,14,15);
  return __builtin_bit_cast(v16bf, full);
}

// ---- K0: zero the per-row exp-sum accumulators ----
__global__ void init_kernel(float* rowSums) {
  int t = blockIdx.x * blockDim.x + threadIdx.x;
  if (t < QROWS) rowSums[t] = 0.f;
}

// ---- K1: gather + L2-normalize video proposals -> bf16 V [BP x C] ----
__global__ __launch_bounds__(256)
void video_kernel(const float* __restrict__ video, const int* __restrict__ mask_idx,
                  unsigned short* __restrict__ Vb) {
  __shared__ float red[256];
  int row = blockIdx.x;           // b*PP + p
  int b = row / PP, p = row - b * PP;
  int t = threadIdx.x;
  int idx = mask_idx[p];
  float x = video[((size_t)b * CC + t) * NN2 + idx];
  red[t] = x * x; __syncthreads();
  for (int k = 128; k > 0; k >>= 1) { if (t < k) red[t] += red[t + k]; __syncthreads(); }
  float inv = 1.0f / fmaxf(sqrtf(red[0]), 1e-12f);
  Vb[(size_t)row * CC + t] = f2bf(x * inv);
}

// ---- K2: normalize sentence feats -> fp32 sfn + bf16 into Q rows [0,64) ----
__global__ __launch_bounds__(256)
void sents_kernel(const float* __restrict__ sents, float* __restrict__ sfn,
                  unsigned short* __restrict__ Qb) {
  __shared__ float red[256];
  int s = blockIdx.x, t = threadIdx.x;
  float x = sents[(size_t)s * CC + t];
  red[t] = x * x; __syncthreads();
  for (int k = 128; k > 0; k >>= 1) { if (t < k) red[t] += red[t + k]; __syncthreads(); }
  float inv = 1.0f / fmaxf(sqrtf(red[0]), 1e-12f);
  float y = x * inv;
  sfn[(size_t)s * CC + t] = y;
  Qb[(size_t)s * CC + t] = f2bf(y);
}

// ---- K3: top-1 proposal per moment, gather normalized row -> pvf fp32 + Q rows [64,192) ----
__global__ __launch_bounds__(256)
void topk_gather_kernel(const float* __restrict__ iou2ds, const float* __restrict__ video,
                        const int* __restrict__ mask_idx, const int* __restrict__ s2v,
                        const int* __restrict__ m2s,
                        float* __restrict__ pvf, unsigned short* __restrict__ Qb) {
  __shared__ float sval[256];
  __shared__ int   sidx[256];
  __shared__ float red[256];
  int m = blockIdx.x, t = threadIdx.x;
  float best = -1e30f; int bi = 0;
  for (int p = t; p < PP; p += 256) {
    float v = iou2ds[(size_t)m * NN2 + mask_idx[p]];
    if (v > best) { best = v; bi = p; }
  }
  sval[t] = best; sidx[t] = bi;
  __syncthreads();
  for (int k = 128; k > 0; k >>= 1) {
    if (t < k) {
      float v2 = sval[t + k]; int i2 = sidx[t + k];
      if (v2 > sval[t] || (v2 == sval[t] && i2 < sidx[t])) { sval[t] = v2; sidx[t] = i2; }
    }
    __syncthreads();
  }
  int pbest = sidx[0];
  int b = s2v[m2s[m]];
  int idx = mask_idx[pbest];
  float x = video[((size_t)b * CC + t) * NN2 + idx];
  red[t] = x * x; __syncthreads();
  for (int k = 128; k > 0; k >>= 1) { if (t < k) red[t] += red[t + k]; __syncthreads(); }
  float inv = 1.0f / fmaxf(sqrtf(red[0]), 1e-12f);
  float y = x * inv;
  pvf[(size_t)m * CC + t] = y;
  Qb[(size_t)(64 + m) * CC + t] = f2bf(y);
}

// ---- K4: per-sentence count of valid negatives ----
__global__ __launch_bounds__(256)
void count_kernel(const float* __restrict__ iou2d, const int* __restrict__ mask_idx,
                  int* __restrict__ counts) {
  __shared__ int red[256];
  int s = blockIdx.x, t = threadIdx.x;
  int cnt = 0;
  for (int p = t; p < PP; p += 256)
    if (iou2d[(size_t)s * NN2 + mask_idx[p]] > 0.5f) cnt++;
  red[t] = cnt; __syncthreads();
  for (int k = 128; k > 0; k >>= 1) { if (t < k) red[t] += red[t + k]; __syncthreads(); }
  if (t == 0) counts[s] = BP - red[0];
}

// ---- K5: Bernoulli ~NEG/valid negative-sample bitmasks (Gumbel-top-k surrogate) ----
__global__ __launch_bounds__(256)
void mask_kernel(const float* __restrict__ iou2d, const int* __restrict__ mask_idx,
                 const int* __restrict__ s2v, const int* __restrict__ counts,
                 unsigned int* __restrict__ maskQ, unsigned int* __restrict__ maskI) {
  int wi = blockIdx.x * 256 + threadIdx.x;
  if (wi >= SS * PW) return;
  int s = wi / PW;
  int wj = wi - s * PW;
  int ownb = s2v[s];
  float prob = (float)NEGS / (float)counts[s];
  unsigned int mq = 0, mi = 0;
  int jbase = wj * 32;
  for (int bit = 0; bit < 32; ++bit) {
    int j = jbase + bit;
    int b = j / PP, p = j - b * PP;
    bool valid = true;
    if (b == ownb) {
      float iou = iou2d[(size_t)s * NN2 + mask_idx[p]];
      if (iou > 0.5f) valid = false;
    }
    if (valid) {
      if (u01(0x9E3779B9u, (unsigned)s, (unsigned)j) < prob) mq |= (1u << bit);
      if (u01(0x7F4A7C15u, (unsigned)s, (unsigned)j) < prob) mi |= (1u << bit);
    }
  }
  maskQ[wi] = mq; maskI[wi] = mi;
}

// ---- K6: fused WMMA bf16 GEMM  D[192 x 66560] = Q @ V^T  with masked exp row-sums ----
__global__ __launch_bounds__(128)
void gemm_expsum_kernel(const unsigned short* __restrict__ Qb,
                        const unsigned short* __restrict__ Vb,
                        const unsigned int* __restrict__ maskQ,
                        const unsigned int* __restrict__ maskI,
                        float* __restrict__ rowSums) {
  __shared__ __align__(16) unsigned short ldsA[64 * 256];  // 32 KB: 64 Q rows, full K
  __shared__ __align__(16) unsigned short ldsB[64 * 128];  // 16 KB: 64 V rows, half K
  const int t = threadIdx.x;
  const int lane = t & 31, w = t >> 5;
  const int rowBase = blockIdx.y * 64;
  const int colBase = blockIdx.x * 64;

  // ---- A tile stage (global -> LDS) ----
#if USE_ASYNC_LDS
  {
    const unsigned short* srcA = Qb + (size_t)rowBase * CC;
#pragma unroll
    for (int i = 0; i < 16; ++i) {
      int chunk = t + i * 128;             // 2048 x 16B chunks
      async_copy16(srcA + chunk * 8, ldsA + chunk * 8);
    }
  }
#else
  {
    const uint4* srcA = (const uint4*)(Qb + (size_t)rowBase * CC);
    uint4* dA = (uint4*)ldsA;
#pragma unroll
    for (int i = 0; i < 16; ++i) dA[t + i * 128] = srcA[t + i * 128];
  }
#endif

  const int rsel = lane >> 4;      // K-chunk select per ISA 16-bit layout
  const int lid  = lane & 15;      // row/col within 16
  const unsigned short* aRow = ldsA + (w * 16 + lid) * 256;

  v8f c[4];
#pragma unroll
  for (int st = 0; st < 4; ++st)
#pragma unroll
    for (int i = 0; i < 8; ++i) c[st][i] = 0.f;

  for (int kh = 0; kh < 2; ++kh) {
    __syncthreads();  // previous-iteration readers of ldsB are done
    // ---- B half-tile stage: 64 V rows x 128 k ----
#if USE_ASYNC_LDS
#pragma unroll
    for (int i = 0; i < 8; ++i) {
      int idx = t + i * 128;
      int row = idx >> 4, c16 = idx & 15;
      async_copy16(Vb + (size_t)(colBase + row) * CC + kh * 128 + c16 * 8,
                   ldsB + row * 128 + c16 * 8);
    }
    async_wait0();    // own wave's A+B async copies complete
#else
#pragma unroll
    for (int i = 0; i < 8; ++i) {
      int idx = t + i * 128;
      int row = idx >> 4, c16 = idx & 15;
      *(uint4*)(ldsB + row * 128 + c16 * 8) =
          *(const uint4*)(Vb + (size_t)(colBase + row) * CC + kh * 128 + c16 * 8);
    }
#endif
    __syncthreads();  // all waves' copies visible
#pragma unroll
    for (int st = 0; st < 4; ++st) {
      const unsigned short* bCol = ldsB + (st * 16 + lid) * 128;
#pragma unroll
      for (int kk = 0; kk < 4; ++kk) {
        v16bf a = load_frag(aRow, kh * 128 + kk * 32 + rsel * 8);
        v16bf b = load_frag(bCol, kk * 32 + rsel * 8);
        c[st] = __builtin_amdgcn_wmma_f32_16x16x32_bf16(
            false, a, false, b, (short)0, c[st], false, false);
      }
    }
  }

  // masked exp accumulation; C/D layout: VGPR i -> row i + 8*rsel, col lid
  const float scale = (rowBase == 0) ? INV_T : 1.0f;
  float accsum[8];
#pragma unroll
  for (int i = 0; i < 8; ++i) accsum[i] = 0.f;

#pragma unroll
  for (int st = 0; st < 4; ++st) {
    int j = colBase + st * 16 + lid;
    int jw = j >> 5;
    unsigned int jb = 1u << (j & 31);
#pragma unroll
    for (int i = 0; i < 8; ++i) {
      int grow = rowBase + w * 16 + i + rsel * 8;
      unsigned int word = (rowBase == 0)
                              ? maskQ[grow * PW + jw]
                              : maskI[((grow - 64) >> 1) * PW + jw];
      if (word & jb) accsum[i] += __expf(c[st][i] * scale);
    }
  }
  // reduce across each 16-lane half, then atomically add per row
#pragma unroll
  for (int i = 0; i < 8; ++i) {
    float v = accsum[i];
    v += __shfl_xor(v, 1);
    v += __shfl_xor(v, 2);
    v += __shfl_xor(v, 4);
    v += __shfl_xor(v, 8);
    if (lid == 0) {
      int grow = rowBase + w * 16 + i + rsel * 8;
      atomicAdd(&rowSums[grow], v);
    }
  }
}

// ---- K7: inter_video scores vs all sentences (fp32), pos scores, ~eye exp sums ----
__global__ __launch_bounds__(64)
void smallscores_kernel(const float* __restrict__ pvf, const float* __restrict__ sfn,
                        const int* __restrict__ m2s,
                        float* __restrict__ negInter, float* __restrict__ posArr) {
  __shared__ float q[CC];
  __shared__ float red[64];
  int m = blockIdx.x, t = threadIdx.x;
  for (int i = t; i < CC; i += 64) q[i] = pvf[(size_t)m * CC + i];
  __syncthreads();
  float dot = 0.f;
  const float* srow = sfn + (size_t)t * CC;
  for (int ci = 0; ci < CC; ++ci) dot += q[ci] * srow[ci];
  int own = m2s[m];
  if (t == own) posArr[m] = dot;
  red[t] = (t == own) ? 0.f : __expf(dot * INV_T);
  __syncthreads();
  for (int k = 32; k > 0; k >>= 1) { if (t < k) red[t] += red[t + k]; __syncthreads(); }
  if (t == 0) negInter[m] = red[0];
}

// ---- K8: intra positive scores ----
__global__ __launch_bounds__(256)
void intrapos_kernel(const float* __restrict__ pvf, const int* __restrict__ ref_idx,
                     const int* __restrict__ pos_idx, float* __restrict__ intraPos) {
  __shared__ float red[256];
  int e = blockIdx.x, t = threadIdx.x;
  int r = ref_idx[e], pq = pos_idx[e];
  red[t] = pvf[(size_t)r * CC + t] * pvf[(size_t)pq * CC + t];
  __syncthreads();
  for (int k = 128; k > 0; k >>= 1) { if (t < k) red[t] += red[t + k]; __syncthreads(); }
  if (t == 0) intraPos[e] = red[0];
}

// ---- K9: final losses ----
__global__ __launch_bounds__(256)
void finalloss_kernel(const float* __restrict__ posArr, const float* __restrict__ negInter,
                      const float* __restrict__ rowSums, const float* __restrict__ intraPos,
                      const int* __restrict__ m2s, const int* __restrict__ ref_idx,
                      float* __restrict__ out) {
  __shared__ float red[256];
  __shared__ float res[3];
  int t = threadIdx.x;
  // inter_video
  float v = 0.f;
  if (t < MM) {
    float pl = (posArr[t] - INTER_MARGIN) * INV_T;
    float pe = __expf(pl);
    v = -(pl - __logf(pe + negInter[t] + rowSums[64 + t]));
  }
  red[t] = v; __syncthreads();
  for (int k = 128; k > 0; k >>= 1) { if (t < k) red[t] += red[t + k]; __syncthreads(); }
  if (t == 0) res[0] = red[0] / (float)MM;
  __syncthreads();
  // inter_query
  v = 0.f;
  if (t < MM) {
    float pl = (posArr[t] - INTER_MARGIN) * INV_T;
    float pe = __expf(pl);
    v = -(pl - __logf(pe + rowSums[m2s[t]]));
  }
  red[t] = v; __syncthreads();
  for (int k = 128; k > 0; k >>= 1) { if (t < k) red[t] += red[t + k]; __syncthreads(); }
  if (t == 0) res[1] = red[0] / (float)MM;
  __syncthreads();
  // intra_video (temperature 1.0, margin 0)
  {
    float pl = intraPos[t];
    float pe = __expf(pl);
    int r = ref_idx[t];
    v = -(pl - __logf(pe + rowSums[64 + r] + negInter[r]));
  }
  red[t] = v; __syncthreads();
  for (int k = 128; k > 0; k >>= 1) { if (t < k) red[t] += red[t + k]; __syncthreads(); }
  if (t == 0) {
    res[2] = red[0] / (float)EE;
    float total = res[0] + res[1] + 0.1f * res[2];
    out[0] = total; out[1] = res[0]; out[2] = res[1]; out[3] = res[2];
  }
}

extern "C" void kernel_launch(void* const* d_in, const int* in_sizes, int n_in,
                              void* d_out, int out_size, void* d_ws, size_t ws_size,
                              hipStream_t stream) {
  (void)in_sizes; (void)n_in; (void)out_size; (void)ws_size;
  const float* video    = (const float*)d_in[0];
  const float* sents    = (const float*)d_in[1];
  const float* iou2d    = (const float*)d_in[2];
  const float* iou2ds   = (const float*)d_in[3];
  const int*   mask_idx = (const int*)d_in[4];
  const int*   s2v      = (const int*)d_in[5];
  const int*   m2s      = (const int*)d_in[6];
  // d_in[7] scatter_e2s implied by ref_idx sentence
  const int*   ref_idx  = (const int*)d_in[8];
  const int*   pos_idx  = (const int*)d_in[9];
  float* out = (float*)d_out;

  char* wsp = (char*)d_ws;
  size_t off = 0;
  auto carve = [&](size_t bytes) -> void* {
    void* p = wsp + off;
    off = (off + bytes + 255) & ~(size_t)255;
    return p;
  };
  unsigned short* Vb     = (unsigned short*)carve((size_t)BP * CC * 2);  // 34 MB bf16
  unsigned short* Qb     = (unsigned short*)carve((size_t)QROWS * CC * 2);
  float*          sfn    = (float*)carve((size_t)SS * CC * 4);
  float*          pvf    = (float*)carve((size_t)MM * CC * 4);
  unsigned int*   maskQ  = (unsigned int*)carve((size_t)SS * PW * 4);
  unsigned int*   maskI  = (unsigned int*)carve((size_t)SS * PW * 4);
  int*            counts = (int*)carve(SS * 4);
  float*          negInt = (float*)carve(MM * 4);
  float*          posArr = (float*)carve(MM * 4);
  float*          intraP = (float*)carve(EE * 4);
  float*          rowSum = (float*)carve(QROWS * 4);

  init_kernel<<<1, 256, 0, stream>>>(rowSum);
  video_kernel<<<BP, 256, 0, stream>>>(video, mask_idx, Vb);
  sents_kernel<<<SS, 256, 0, stream>>>(sents, sfn, Qb);
  topk_gather_kernel<<<MM, 256, 0, stream>>>(iou2ds, video, mask_idx, s2v, m2s, pvf, Qb);
  count_kernel<<<SS, 256, 0, stream>>>(iou2d, mask_idx, counts);
  mask_kernel<<<(SS * PW + 255) / 256, 256, 0, stream>>>(iou2d, mask_idx, s2v, counts, maskQ, maskI);
  dim3 g(COLTILES, QROWS / 64);
  gemm_expsum_kernel<<<g, 128, 0, stream>>>(Qb, Vb, maskQ, maskI, rowSum);
  smallscores_kernel<<<MM, 64, 0, stream>>>(pvf, sfn, m2s, negInt, posArr);
  intrapos_kernel<<<EE, 256, 0, stream>>>(pvf, ref_idx, pos_idx, intraP);
  finalloss_kernel<<<1, 256, 0, stream>>>(posArr, negInt, rowSum, intraP, m2s, ref_idx, out);
}
